// MyLSTM_AM_81896436400415
// MI455X (gfx1250) — matrix operand order repaired and checked by reading, not compiled, not run on hardware
//
#include <hip/hip_runtime.h>
#include <hip/hip_bf16.h>

// ---------------------------------------------------------------------------
// Fused encoder-LSTM + attention-decoder + MLP head for MI455X (gfx1250).
// One workgroup (256 thr = 8 waves, wave32) owns a 16-row batch tile and
// carries h/c through all 160 sequential steps. All matmuls use
// v_wmma_f32_16x16x32_bf16 (fp32 accumulate), weights pre-packed to bf16.
// ---------------------------------------------------------------------------

typedef __attribute__((ext_vector_type(16))) __bf16 v16bf;
typedef __attribute__((ext_vector_type(8)))  float  v8f;

#define BSZ   512
#define TLEN  128
#define INP   64
#define HID   512
#define G4    2048           // 4*HID gate width
#define TSL   32             // attention buffer slots
#define PL    32             // decoder length
#define KENC  576            // HID + INP combined K for encoder GEMM

union BF16Frag { v16bf v; unsigned int u[8]; unsigned short s[16]; };

__device__ inline unsigned short f2bf(float x) {
  unsigned int u = __float_as_uint(x);
  u += 0x7FFFu + ((u >> 16) & 1u);          // round-to-nearest-even
  return (unsigned short)(u >> 16);
}
__device__ inline float bf2f(unsigned short s) {
  return __uint_as_float(((unsigned int)s) << 16);
}
__device__ inline float sigmoidf_(float x) { return 1.0f / (1.0f + __expf(-x)); }

// ---- 4x4-tile gate GEMM: acc[blk][j] += A(16xK) * W(cols blk*512+wbase+j*16)
// A: LDS row-major bf16, stride `astride`. W: global row-major bf16 [G4][ldk].
__device__ inline void gates_mma(v8f (&acc)[4][4], const unsigned short* Abase,
                                 int astride, const unsigned short* W, int ldk,
                                 int ktn, int wbase, int lane) {
  const int rowm = lane & 15;
  const int hi   = lane >> 4;              // 0 / 1
  const unsigned short* arow = Abase + rowm * astride;
#pragma unroll 1
  for (int kt = 0; kt < ktn; ++kt) {
    BF16Frag af;
    const unsigned int* pa = (const unsigned int*)(arow + kt * 32 + hi * 16);
    const unsigned int* pb = (const unsigned int*)(arow + kt * 32 + hi * 16 + 8);
#pragma unroll
    for (int i = 0; i < 4; ++i) { af.u[i] = pa[i]; af.u[4 + i] = pb[i]; }
#pragma unroll
    for (int blk = 0; blk < 4; ++blk) {
#pragma unroll
      for (int j = 0; j < 4; ++j) {
        const int col = blk * 512 + wbase + j * 16 + (lane & 15);
        BF16Frag bf;
        const unsigned int* pw =
            (const unsigned int*)(W + (size_t)col * ldk + kt * 32 + hi * 16);
#pragma unroll
        for (int i = 0; i < 8; ++i) bf.u[i] = pw[i];
        acc[blk][j] = __builtin_amdgcn_wmma_f32_16x16x32_bf16(
            false, af.v, false, bf.v, (short)0, acc[blk][j], false, false);
      }
    }
  }
}

// ---- head GEMM: acc[j] += A(16x1024) * W1(cols wbase+j*16), K=1024
__device__ inline void head_mma(v8f (&acc)[4], const unsigned short* Abase,
                                const unsigned short* W1, int wbase, int lane) {
  const int rowm = lane & 15;
  const int hi   = lane >> 4;
  const unsigned short* arow = Abase + rowm * 1024;
#pragma unroll 1
  for (int kt = 0; kt < 32; ++kt) {
    BF16Frag af;
    const unsigned int* pa = (const unsigned int*)(arow + kt * 32 + hi * 16);
    const unsigned int* pb = (const unsigned int*)(arow + kt * 32 + hi * 16 + 8);
#pragma unroll
    for (int i = 0; i < 4; ++i) { af.u[i] = pa[i]; af.u[4 + i] = pb[i]; }
#pragma unroll
    for (int j = 0; j < 4; ++j) {
      const int col = wbase + j * 16 + (lane & 15);
      BF16Frag bf;
      const unsigned int* pw =
          (const unsigned int*)(W1 + (size_t)col * 1024 + kt * 32 + hi * 16);
#pragma unroll
      for (int i = 0; i < 8; ++i) bf.u[i] = pw[i];
      acc[j] = __builtin_amdgcn_wmma_f32_16x16x32_bf16(
          false, af.v, false, bf.v, (short)0, acc[j], false, false);
    }
  }
}

// ---------------- weight prep: fp32 -> bf16 packs ----------------
__global__ void prep_enc_k(const float* __restrict__ W_hh,
                           const float* __restrict__ W_ih,
                           unsigned short* __restrict__ Wenc) {
  int i = blockIdx.x * blockDim.x + threadIdx.x;
  if (i >= G4 * KENC) return;
  int g = i / KENC, k = i % KENC;
  float v = (k < HID) ? W_hh[(size_t)g * HID + k] : W_ih[(size_t)g * INP + (k - HID)];
  Wenc[i] = f2bf(v);
}
__global__ void prep_wc_k(const float* __restrict__ Wc_ih,
                          const float* __restrict__ Wc_hh,
                          unsigned short* __restrict__ Wc) {
  int i = blockIdx.x * blockDim.x + threadIdx.x;
  if (i >= G4 * HID) return;
  Wc[i] = f2bf(Wc_ih[i] + Wc_hh[i]);
}
__global__ void prep_w1_k(const float* __restrict__ W1,
                          unsigned short* __restrict__ W1b) {
  int i = blockIdx.x * blockDim.x + threadIdx.x;
  if (i >= HID * 2 * HID) return;
  W1b[i] = f2bf(W1[i]);
}

// ---------------- fused persistent kernel ----------------
__global__ __launch_bounds__(256) void fused_lstm_attn_k(
    const float* __restrict__ x, const float* __restrict__ b_lstm,
    const float* __restrict__ b_cell, const float* __restrict__ b1,
    const float* __restrict__ W2, const float* __restrict__ b2,
    const unsigned short* __restrict__ Wenc, const unsigned short* __restrict__ Wc,
    const unsigned short* __restrict__ W1b, unsigned short* __restrict__ buf,
    float* __restrict__ out) {
  __shared__ unsigned short a_bf[16][KENC];    // [h ; x_t] A-matrix, bf16
  __shared__ unsigned short out_bf[16][1024];  // [ctx ; h_t] for head GEMM
  __shared__ float scores_l[16][TSL];
  __shared__ float ybuf[16];

  const int tid   = threadIdx.x;
  const int wv    = tid >> 5;
  const int lane  = tid & 31;
  const int n     = lane & 15;
  const int hi    = lane >> 4;
  const int wbase = wv * 64;                   // 64 hidden units per wave
  const int bBase = blockIdx.x * 16;

  // per-lane constant caches
  float biasE[16], biasD[16], b1c[4], w2c[4];
#pragma unroll
  for (int blk = 0; blk < 4; ++blk)
#pragma unroll
    for (int j = 0; j < 4; ++j) {
      int col = blk * 512 + wbase + j * 16 + n;
      biasE[blk * 4 + j] = b_lstm[col];
      biasD[blk * 4 + j] = b_cell[col];
    }
#pragma unroll
  for (int j = 0; j < 4; ++j) {
    int col = wbase + j * 16 + n;
    b1c[j] = b1[col];
    w2c[j] = W2[col];
  }
  const float b2v = b2[0];

  float creg[4][8];
#pragma unroll
  for (int j = 0; j < 4; ++j)
#pragma unroll
    for (int r = 0; r < 8; ++r) creg[j][r] = 0.0f;

  // zero h region of A
  for (int i = tid; i < 16 * HID; i += 256) a_bf[i >> 9][i & 511] = 0;

  // ================= encoder: 128 sequential steps =================
#pragma unroll 1
  for (int t = 0; t < TLEN; ++t) {
    // stage x_t tile (bf16) into A cols [512,576)
    for (int i = tid; i < 16 * INP; i += 256) {
      int r = i >> 6, c = i & 63;
      a_bf[r][HID + c] = f2bf(x[((size_t)(bBase + r) * TLEN + t) * INP + c]);
    }
    __syncthreads();                       // x + prev h visible

    v8f acc[4][4];
#pragma unroll
    for (int bq = 0; bq < 4; ++bq)
#pragma unroll
      for (int j = 0; j < 4; ++j) acc[bq][j] = (v8f)0.0f;
    gates_mma(acc, &a_bf[0][0], KENC, Wenc, KENC, KENC / 32, wbase, lane);
    __syncthreads();                       // reads done before h overwrite

#pragma unroll
    for (int j = 0; j < 4; ++j) {
      const int hid = wbase + j * 16 + n;
#pragma unroll
      for (int r = 0; r < 8; ++r) {
        float iv = sigmoidf_(acc[0][j][r] + biasE[j]);
        float fv = sigmoidf_(acc[1][j][r] + biasE[4 + j]);
        float gv = tanhf(acc[2][j][r] + biasE[8 + j]);
        float ov = sigmoidf_(acc[3][j][r] + biasE[12 + j]);
        float cn = fv * creg[j][r] + iv * gv;
        creg[j][r] = cn;
        unsigned short hb = f2bf(ov * tanhf(cn));
        int row = hi * 8 + r;
        a_bf[row][hid] = hb;
        if (t >= TLEN - TSL)               // seed attention buffer
          buf[((size_t)(bBase + row) * TSL + (t - (TLEN - TSL))) * HID + hid] = hb;
      }
    }
  }

  // ================= decoder: 32 steps w/ attention + head =================
#pragma unroll 1
  for (int d = 0; d < PL; ++d) {
    __syncthreads();                       // h visible / ybuf consumed

    v8f acc[4][4];
#pragma unroll
    for (int bq = 0; bq < 4; ++bq)
#pragma unroll
      for (int j = 0; j < 4; ++j) acc[bq][j] = (v8f)0.0f;
    gates_mma(acc, &a_bf[0][0], KENC, Wc, HID, HID / 32, wbase, lane);
    __syncthreads();                       // reads done

#pragma unroll
    for (int j = 0; j < 4; ++j) {
      const int hid = wbase + j * 16 + n;
#pragma unroll
      for (int r = 0; r < 8; ++r) {
        float iv = sigmoidf_(acc[0][j][r] + biasD[j]);
        float fv = sigmoidf_(acc[1][j][r] + biasD[4 + j]);
        float gv = tanhf(acc[2][j][r] + biasD[8 + j]);
        float ov = sigmoidf_(acc[3][j][r] + biasD[12 + j]);
        float cn = fv * creg[j][r] + iv * gv;
        creg[j][r] = cn;
        a_bf[hi * 8 + r][hid] = f2bf(ov * tanhf(cn));
      }
    }
    __syncthreads();                       // h_t visible

    // attention scores: h_t . buf[s]   (16 rows x 32 slots)
    for (int e = tid; e < 16 * TSL; e += 256) {
      int row = e >> 5, s = e & 31;
      const unsigned short* bp = buf + ((size_t)(bBase + row) * TSL + s) * HID;
      float a = 0.0f;
      for (int k = 0; k < HID; ++k) a += bf2f(a_bf[row][k]) * bf2f(bp[k]);
      scores_l[row][s] = a;
    }
    if (tid < 16) ybuf[tid] = b2v;
    __syncthreads();

    if (tid < 16) {                        // softmax per row
      float m = scores_l[tid][0];
#pragma unroll
      for (int s = 1; s < TSL; ++s) m = fmaxf(m, scores_l[tid][s]);
      float sum = 0.0f;
#pragma unroll
      for (int s = 0; s < TSL; ++s) { float e = __expf(scores_l[tid][s] - m); scores_l[tid][s] = e; sum += e; }
      float inv = 1.0f / sum;
#pragma unroll
      for (int s = 0; s < TSL; ++s) scores_l[tid][s] *= inv;
    }
    __syncthreads();

    // context + concat [ctx ; h_t] -> out_bf
    for (int e = tid; e < 16 * HID; e += 256) {
      int row = e >> 9, k = e & 511;
      float c = 0.0f;
      const unsigned short* bp = buf + (size_t)(bBase + row) * TSL * HID + k;
#pragma unroll 1
      for (int s = 0; s < TSL; ++s) c += scores_l[row][s] * bf2f(bp[(size_t)s * HID]);
      out_bf[row][k] = f2bf(c);
      out_bf[row][HID + k] = a_bf[row][k];
    }
    __syncthreads();                       // out_bf ready, buf reads done

    // rotate h_t into buffer slot d (permutation-invariant circular buffer)
    for (int e = tid; e < 16 * HID; e += 256) {
      int row = e >> 9, k = e & 511;
      buf[((size_t)(bBase + row) * TSL + d) * HID + k] = a_bf[row][k];
    }

    // head: y = tanh([ctx;h] W1^T + b1) . W2 + b2   via WMMA + LDS atomics
    v8f acc2[4];
#pragma unroll
    for (int j = 0; j < 4; ++j) acc2[j] = (v8f)0.0f;
    head_mma(acc2, &out_bf[0][0], W1b, wbase, lane);

    float sacc[8];
#pragma unroll
    for (int r = 0; r < 8; ++r) sacc[r] = 0.0f;
#pragma unroll
    for (int j = 0; j < 4; ++j)
#pragma unroll
      for (int r = 0; r < 8; ++r)
        sacc[r] += tanhf(acc2[j][r] + b1c[j]) * w2c[j];
#pragma unroll
    for (int r = 0; r < 8; ++r) atomicAdd(&ybuf[hi * 8 + r], sacc[r]);
    __syncthreads();

    if (tid < 16) out[(size_t)(bBase + tid) * PL + d] = ybuf[tid];
  }
}

extern "C" void kernel_launch(void* const* d_in, const int* in_sizes, int n_in,
                              void* d_out, int out_size, void* d_ws, size_t ws_size,
                              hipStream_t stream) {
  const float* x      = (const float*)d_in[0];
  const float* W_ih   = (const float*)d_in[1];
  const float* W_hh   = (const float*)d_in[2];
  const float* b_lstm = (const float*)d_in[3];
  const float* Wc_ih  = (const float*)d_in[4];
  const float* Wc_hh  = (const float*)d_in[5];
  const float* b_cell = (const float*)d_in[6];
  const float* W1     = (const float*)d_in[7];
  const float* b1     = (const float*)d_in[8];
  const float* W2     = (const float*)d_in[9];
  const float* b2     = (const float*)d_in[10];
  float* out = (float*)d_out;

  // workspace layout (bf16): Wenc[2048][576] | Wc[2048][512] | W1b[512][1024] | buf[512][32][512]
  unsigned short* Wenc = (unsigned short*)d_ws;
  unsigned short* Wc   = Wenc + (size_t)G4 * KENC;
  unsigned short* W1b  = Wc   + (size_t)G4 * HID;
  unsigned short* buf  = W1b  + (size_t)HID * 2 * HID;

  prep_enc_k<<<(G4 * KENC + 255) / 256, 256, 0, stream>>>(W_hh, W_ih, Wenc);
  prep_wc_k <<<(G4 * HID + 255) / 256, 256, 0, stream>>>(Wc_ih, Wc_hh, Wc);
  prep_w1_k <<<(HID * 2 * HID + 255) / 256, 256, 0, stream>>>(W1, W1b);

  fused_lstm_attn_k<<<BSZ / 16, 256, 0, stream>>>(
      x, b_lstm, b_cell, b1, W2, b2, Wenc, Wc, W1b, buf, out);
}
